// SlidingWindowAttention_10900626997690
// MI455X (gfx1250) — matrix operand order repaired and checked by reading, not compiled, not run on hardware
//
#include <hip/hip_runtime.h>

// ---------------------------------------------------------------------------
// Sliding-window attention for MI455X (gfx1250), bf16 WMMA pipeline.
//   Stage 1: qkv = x @ Wqkv^T          (WMMA bf16, f32 accum) -> Q,K,V^T (bf16, ws)
//   Stage 2: flash-style sliding-window attention, WG-shared K/V tiles in LDS
//            (async global->LDS copies when available), WMMA QK^T and P@V
//   Stage 3: out = attn @ Wo^T          (WMMA bf16, f32 accum) -> f32 d_out
// ---------------------------------------------------------------------------

typedef __attribute__((ext_vector_type(16))) __bf16 bf16x16;
typedef __attribute__((ext_vector_type(8)))  __bf16 bf16x8;
typedef __attribute__((ext_vector_type(8)))  float  f32x8;
typedef __attribute__((ext_vector_type(4)))  float  f32x4;

union BF16Frag { bf16x16 v; bf16x8 h[2]; };

constexpr int D_MODEL  = 1024;
constexpr int NHEADS   = 16;
constexpr int HDIM     = 64;
constexpr int SEQ      = 2048;
constexpr int BATCH    = 4;
constexpr int HALF_WIN = 256;
constexpr float INV_SCALE = 0.125f;  // 1/sqrt(64)

// ws layout (bf16 elements):
//   Q  : [64][2048][64]   (bh, s, d)      offset 0
//   K  : [64][2048][64]   (bh, s, d)      offset QK_ELEMS
//   V^T: [64][64][2048]   (bh, d, s)      offset 2*QK_ELEMS
//   AO : [8192][1024]     (b*s, h*d)      offset 3*QK_ELEMS
constexpr size_t QK_ELEMS = (size_t)64 * 2048 * 64;  // 8,388,608 elems

// ---- gfx1250 async global->LDS copy (GLOBAL_LOAD_ASYNC_TO_LDS_B128) --------
#if defined(__has_builtin)
#if __has_builtin(__builtin_amdgcn_global_load_async_to_lds_b128) && \
    __has_builtin(__builtin_amdgcn_s_wait_asynccnt)
#define HAS_ASYNC 1
#endif
#endif
#ifndef HAS_ASYNC
#define HAS_ASYNC 0
#endif

#if HAS_ASYNC
// Builtin prototype (from hipcc diagnostic): param 0 is
//   'int __attribute__((vector_size(16))) __device__ *'  (AS(1) int4*)
typedef int v4i_vs __attribute__((vector_size(4 * sizeof(int))));
typedef __attribute__((address_space(1))) v4i_vs g_v4i;
typedef __attribute__((address_space(3))) v4i_vs l_v4i;
static __device__ __forceinline__ void cp16_async(const __bf16* g, __bf16* l) {
  // generic LDS pointer low 32 bits == LDS offset (ISA aperture rule)
  g_v4i* gp = (g_v4i*)(__UINTPTR_TYPE__)g;
  l_v4i* lp = (l_v4i*)(unsigned int)(__UINTPTR_TYPE__)l;
  __builtin_amdgcn_global_load_async_to_lds_b128(gp, lp, 0, 0);
}
static __device__ __forceinline__ void async_drain() {
  __builtin_amdgcn_s_wait_asynccnt(0);
}
#else
static __device__ __forceinline__ void cp16_async(const __bf16* g, __bf16* l) {
  *(bf16x8*)l = *(const bf16x8*)g;   // sync fallback through VGPRs
}
static __device__ __forceinline__ void async_drain() {}
#endif

static __device__ __forceinline__ void cp32(const __bf16* g, __bf16* l) {
  cp16_async(g, l);
  cp16_async(g + 8, l + 8);
}

static __device__ __forceinline__ f32x8 zero8() {
  f32x8 z = {0.f, 0.f, 0.f, 0.f, 0.f, 0.f, 0.f, 0.f};
  return z;
}

// ---------------------------------------------------------------------------
// Tiled GEMM:  C[M,N] = A[M,K] @ W[N,K]^T
//   MODE 0: A = f32 x, epilogue scatters bf16 into Q / K / V^T regions of ws
//   MODE 1: A = bf16 attn-out (async-staged), epilogue writes f32 to d_out
// WG = 256 threads (8 waves), tile 128x128, wave sub-tile 64x32.
// ---------------------------------------------------------------------------
template<int MODE>
__global__ __launch_bounds__(256) void gemm128(
    const float* __restrict__ Af, const __bf16* __restrict__ Ab,
    const float* __restrict__ W,
    float* __restrict__ outF, __bf16* __restrict__ wsb,
    int M, int N, int Kdim)
{
  __shared__ __bf16 As[128][32];
  __shared__ __bf16 Bs[128][32];

  const int t      = threadIdx.x;
  const int lane   = t & 31;
  const int wid    = t >> 5;
  const int wave_m = wid >> 2;     // 0..1  -> 64 rows each
  const int wave_n = wid & 3;      // 0..3  -> 32 cols each
  const int m0 = blockIdx.y * 128;
  const int n0 = blockIdx.x * 128;

  const int srow = t >> 1;          // staging row 0..127
  const int scol = (t & 1) * 16;    // staging col 0 / 16

  f32x8 acc[4][2];
#pragma unroll
  for (int i = 0; i < 4; ++i)
#pragma unroll
    for (int j = 0; j < 2; ++j) acc[i][j] = zero8();

  const int mrow = lane & 15;
  const int aoff = (lane < 16) ? 0 : 8;    // A-frag K chunk base
  const int boff = (lane < 16) ? 0 : 8;    // B-frag K chunk base

  for (int k0 = 0; k0 < Kdim; k0 += 32) {
    // ---- stage A tile ----
    if (MODE == 0) {
      const float* src = Af + (size_t)(m0 + srow) * Kdim + k0 + scol;
      __builtin_prefetch(src + 32, 0, 1);
      f32x4 v0 = *(const f32x4*)(src + 0);
      f32x4 v1 = *(const f32x4*)(src + 4);
      f32x4 v2 = *(const f32x4*)(src + 8);
      f32x4 v3 = *(const f32x4*)(src + 12);
      bf16x8 p0 = {(__bf16)v0.x, (__bf16)v0.y, (__bf16)v0.z, (__bf16)v0.w,
                   (__bf16)v1.x, (__bf16)v1.y, (__bf16)v1.z, (__bf16)v1.w};
      bf16x8 p1 = {(__bf16)v2.x, (__bf16)v2.y, (__bf16)v2.z, (__bf16)v2.w,
                   (__bf16)v3.x, (__bf16)v3.y, (__bf16)v3.z, (__bf16)v3.w};
      *(bf16x8*)&As[srow][scol + 0] = p0;
      *(bf16x8*)&As[srow][scol + 8] = p1;
    } else {
      const __bf16* src = Ab + (size_t)(m0 + srow) * Kdim + k0 + scol;
      cp32(src, &As[srow][scol]);   // async global->LDS when available
    }
    // ---- stage W tile (f32 -> bf16) ----
    {
      const float* src = W + (size_t)(n0 + srow) * Kdim + k0 + scol;
      __builtin_prefetch(src + 32, 0, 1);
      f32x4 v0 = *(const f32x4*)(src + 0);
      f32x4 v1 = *(const f32x4*)(src + 4);
      f32x4 v2 = *(const f32x4*)(src + 8);
      f32x4 v3 = *(const f32x4*)(src + 12);
      bf16x8 p0 = {(__bf16)v0.x, (__bf16)v0.y, (__bf16)v0.z, (__bf16)v0.w,
                   (__bf16)v1.x, (__bf16)v1.y, (__bf16)v1.z, (__bf16)v1.w};
      bf16x8 p1 = {(__bf16)v2.x, (__bf16)v2.y, (__bf16)v2.z, (__bf16)v2.w,
                   (__bf16)v3.x, (__bf16)v3.y, (__bf16)v3.z, (__bf16)v3.w};
      *(bf16x8*)&Bs[srow][scol + 0] = p0;
      *(bf16x8*)&Bs[srow][scol + 8] = p1;
    }
    if (MODE == 1) async_drain();
    __syncthreads();

    // ---- fragments in ISA per-lane layouts ----
    BF16Frag afr[4], bfr[2];
#pragma unroll
    for (int mi = 0; mi < 4; ++mi) {
      const __bf16* p = &As[wave_m * 64 + mi * 16 + mrow][0];
      afr[mi].h[0] = *(const bf16x8*)(p + aoff);
      afr[mi].h[1] = *(const bf16x8*)(p + aoff + 16);
    }
#pragma unroll
    for (int ni = 0; ni < 2; ++ni) {
      const __bf16* p = &Bs[wave_n * 32 + ni * 16 + mrow][0];
      bfr[ni].h[0] = *(const bf16x8*)(p + boff);
      bfr[ni].h[1] = *(const bf16x8*)(p + boff + 16);
    }
#pragma unroll
    for (int mi = 0; mi < 4; ++mi)
#pragma unroll
      for (int ni = 0; ni < 2; ++ni)
        acc[mi][ni] = __builtin_amdgcn_wmma_f32_16x16x32_bf16(
            false, afr[mi].v, false, bfr[ni].v, (short)0, acc[mi][ni],
            false, false);
    __syncthreads();
  }

  // ---- epilogue ----
  const int half8 = (lane < 16) ? 0 : 8;
#pragma unroll
  for (int mi = 0; mi < 4; ++mi) {
#pragma unroll
    for (int ni = 0; ni < 2; ++ni) {
#pragma unroll
      for (int e = 0; e < 8; ++e) {
        const int gm = m0 + wave_m * 64 + mi * 16 + e + half8;
        const int gn = n0 + wave_n * 32 + ni * 16 + mrow;
        const float val = acc[mi][ni][e];
        if (MODE == 0) {
          const int b = gm >> 11, s = gm & 2047;
          const int three = gn >> 10, h = (gn >> 6) & 15, d = gn & 63;
          const int bh = b * 16 + h;
          size_t idx;
          if (three == 2)  // V stored transposed: [bh][d][s]
            idx = 2 * QK_ELEMS + ((size_t)bh * 64 + d) * 2048 + s;
          else             // Q / K: [bh][s][d]
            idx = (size_t)three * QK_ELEMS + ((size_t)bh * 2048 + s) * 64 + d;
          wsb[idx] = (__bf16)val;
        } else {
          outF[(size_t)gm * N + gn] = val;
        }
      }
    }
  }
}

// ---------------------------------------------------------------------------
// Sliding-window flash attention.
// WG = 4 waves = 64 queries; K/V^T key tiles (32 keys) staged in LDS, shared
// by all waves, double-buffered with async global->LDS copies.
// ---------------------------------------------------------------------------
__global__ __launch_bounds__(128) void swattn(
    const __bf16* __restrict__ Qb, const __bf16* __restrict__ Kb,
    const __bf16* __restrict__ Vt, __bf16* __restrict__ Ob)
{
  __shared__ __bf16 Ks[2][32][64];   // [buf][key][d]
  __shared__ __bf16 Vs[2][64][32];   // [buf][d][key]   (V^T tile)
  __shared__ __bf16 Ps[4][16][32];   // per-wave P re-layout buffer

  const int t    = threadIdx.x;
  const int lane = t & 31;
  const int w    = t >> 5;
  const int bh   = blockIdx.y;                    // 0..63
  const int q0   = blockIdx.x * 64;               // WG query base
  const int qw   = q0 + w * 16;                   // wave query base

  const __bf16* Q  = Qb + (size_t)bh * SEQ * HDIM;
  const __bf16* Kh = Kb + (size_t)bh * SEQ * HDIM;
  const __bf16* Vh = Vt + (size_t)bh * HDIM * SEQ;

  const int mrow  = lane & 15;
  const int aoff  = (lane < 16) ? 0 : 8;
  const int boff  = (lane < 16) ? 0 : 8;
  const int half8 = (lane < 16) ? 0 : 8;

  // Q fragments (two 32-wide halves of hd=64), resident for whole loop.
  BF16Frag aQ[2];
  {
    const __bf16* qr = Q + (size_t)(qw + mrow) * HDIM;
#pragma unroll
    for (int dh = 0; dh < 2; ++dh) {
      aQ[dh].h[0] = *(const bf16x8*)(qr + dh * 32 + aoff);
      aQ[dh].h[1] = *(const bf16x8*)(qr + dh * 32 + aoff + 16);
    }
  }

  f32x8 accO[4];
#pragma unroll
  for (int i = 0; i < 4; ++i) accO[i] = zero8();
  float mrun[8], lrun[8];
#pragma unroll
  for (int e = 0; e < 8; ++e) { mrun[e] = -3.0e38f; lrun[e] = 0.f; }

  // WG-level key range: union of all 4 waves' windows; multiples of 32.
  int kt0 = q0 - HALF_WIN; if (kt0 < 0) kt0 = 0;
  int khi = q0 + 64 + HALF_WIN; if (khi > SEQ) khi = SEQ;

  // staging maps (128 threads):  K: 32x64, V^T: 64x32 (bf16)
  const int krow = t >> 2, kcol = (t & 3) * 16;
  const int vrow = t >> 1, vcol = (t & 1) * 16;

  // prologue: stage first tile into buffer 0
  cp32(Kh + (size_t)(kt0 + krow) * HDIM + kcol, &Ks[0][krow][kcol]);
  cp32(Vh + (size_t)vrow * SEQ + kt0 + vcol,    &Vs[0][vrow][vcol]);
  async_drain();
  __syncthreads();

  int buf = 0;
  for (int kt = kt0; kt < khi; kt += 32, buf ^= 1) {
    // stage next tile into the other buffer (overlaps with compute)
    if (kt + 32 < khi) {
      const int ktn = kt + 32;
      cp32(Kh + (size_t)(ktn + krow) * HDIM + kcol, &Ks[buf ^ 1][krow][kcol]);
      cp32(Vh + (size_t)vrow * SEQ + ktn + vcol,    &Vs[buf ^ 1][vrow][vcol]);
    }

    // ---- scores S = Q @ K^T for 32 keys (two 16-wide N subtiles) ----
    f32x8 sc[2];
#pragma unroll
    for (int n = 0; n < 2; ++n) {
      f32x8 s = zero8();
#pragma unroll
      for (int dh = 0; dh < 2; ++dh) {
        BF16Frag bK;
        const __bf16* kr = &Ks[buf][n * 16 + mrow][dh * 32 + boff];
        bK.h[0] = *(const bf16x8*)(kr);
        bK.h[1] = *(const bf16x8*)(kr + 16);
        s = __builtin_amdgcn_wmma_f32_16x16x32_bf16(
            false, aQ[dh].v, false, bK.v, (short)0, s, false, false);
      }
      sc[n] = s;
    }

    // ---- mask |q-key| <= 256, scale ----
#pragma unroll
    for (int n = 0; n < 2; ++n) {
      const int key = kt + n * 16 + mrow;
#pragma unroll
      for (int e = 0; e < 8; ++e) {
        const int q = qw + e + half8;
        int dlt = q - key; if (dlt < 0) dlt = -dlt;
        sc[n][e] = (dlt <= HALF_WIN) ? sc[n][e] * INV_SCALE : -3.0e38f;
      }
    }

    // ---- online softmax update (per-row stats; 16-lane butterflies).
    // p forced to exact 0 on masked entries: a wave may see fully-masked
    // tiles (WG-union key range), where exp(s - mnew) would be exp(0).
    float corr[8];
#pragma unroll
    for (int e = 0; e < 8; ++e) {
      float tmax = fmaxf(sc[0][e], sc[1][e]);
#pragma unroll
      for (int off = 8; off >= 1; off >>= 1)
        tmax = fmaxf(tmax, __shfl_xor(tmax, off, 32));
      const float mnew = fmaxf(mrun[e], tmax);
      const float c  = __expf(mrun[e] - mnew);
      const float p0 = (sc[0][e] > -1.0e37f) ? __expf(sc[0][e] - mnew) : 0.f;
      const float p1 = (sc[1][e] > -1.0e37f) ? __expf(sc[1][e] - mnew) : 0.f;
      sc[0][e] = p0; sc[1][e] = p1;
      float rs = p0 + p1;
#pragma unroll
      for (int off = 8; off >= 1; off >>= 1)
        rs += __shfl_xor(rs, off, 32);
      lrun[e] = lrun[e] * c + rs;
      mrun[e] = mnew;
      corr[e] = c;
    }
#pragma unroll
    for (int i = 0; i < 4; ++i)
#pragma unroll
      for (int e = 0; e < 8; ++e) accO[i][e] *= corr[e];

    // ---- P: C-layout -> A-layout via per-wave LDS ----
#pragma unroll
    for (int n = 0; n < 2; ++n)
#pragma unroll
      for (int e = 0; e < 8; ++e)
        Ps[w][e + half8][n * 16 + mrow] = (__bf16)sc[n][e];
    BF16Frag aP;
    {
      const __bf16* pr = &Ps[w][mrow][0];
      aP.h[0] = *(const bf16x8*)(pr + aoff);
      aP.h[1] = *(const bf16x8*)(pr + aoff + 16);
    }

    // ---- accO += P @ V   (V^T tile: contiguous B-frag loads from LDS) ----
#pragma unroll
    for (int dt = 0; dt < 4; ++dt) {
      BF16Frag bV;
      const __bf16* vr = &Vs[buf][dt * 16 + mrow][boff];
      bV.h[0] = *(const bf16x8*)(vr);
      bV.h[1] = *(const bf16x8*)(vr + 16);
      accO[dt] = __builtin_amdgcn_wmma_f32_16x16x32_bf16(
          false, aP.v, false, bV.v, (short)0, accO[dt], false, false);
    }

    async_drain();       // next tile's async copies complete (per-wave)
    __syncthreads();     // all waves done reading/staging before swap
  }

  // ---- normalize + store bf16 attn-out [b*s][h*64+d] ----
  const int b = bh >> 4, h = bh & 15;
  float inv[8];
#pragma unroll
  for (int e = 0; e < 8; ++e) inv[e] = 1.f / lrun[e];
#pragma unroll
  for (int dt = 0; dt < 4; ++dt)
#pragma unroll
    for (int e = 0; e < 8; ++e) {
      const int gm = b * SEQ + qw + e + half8;
      const int gn = h * 64 + dt * 16 + mrow;
      Ob[(size_t)gm * D_MODEL + gn] = (__bf16)(accO[dt][e] * inv[e]);
    }
}

// ---------------------------------------------------------------------------
extern "C" void kernel_launch(void* const* d_in, const int* in_sizes, int n_in,
                              void* d_out, int out_size, void* d_ws, size_t ws_size,
                              hipStream_t stream) {
  (void)in_sizes; (void)n_in; (void)out_size; (void)ws_size;
  const float* x    = (const float*)d_in[0];   // [4,2048,1024]
  const float* Wqkv = (const float*)d_in[1];   // [3072,1024]
  const float* Wo   = (const float*)d_in[2];   // [1024,1024]
  float* out = (float*)d_out;                  // [4,2048,1024]

  __bf16* wsb = (__bf16*)d_ws;
  __bf16* Qb  = wsb;
  __bf16* Kb  = wsb + QK_ELEMS;
  __bf16* Vt  = wsb + 2 * QK_ELEMS;
  __bf16* Ao  = wsb + 3 * QK_ELEMS;

  const int M = BATCH * SEQ;  // 8192

  // 1) QKV projection: [8192,1024] @ [3072,1024]^T -> Q/K/V^T (bf16)
  dim3 g1(3 * D_MODEL / 128, M / 128);
  gemm128<0><<<g1, 256, 0, stream>>>(x, nullptr, Wqkv, nullptr, wsb,
                                     M, 3 * D_MODEL, D_MODEL);

  // 2) sliding-window attention (WG-shared LDS K/V tiles, double-buffered)
  dim3 g2(SEQ / 64, BATCH * NHEADS);
  swattn<<<g2, 128, 0, stream>>>(Qb, Kb, Vt, Ao);

  // 3) output projection: [8192,1024](bf16) @ [1024,1024]^T -> f32 out
  dim3 g3(D_MODEL / 128, M / 128);
  gemm128<1><<<g3, 256, 0, stream>>>(nullptr, Ao, Wo, out, nullptr,
                                     M, D_MODEL, D_MODEL);
}
